// AttentionHead_61426622267717
// MI455X (gfx1250) — matrix operand order repaired and checked by reading, not compiled, not run on hardware
//
#include <hip/hip_runtime.h>
#include <hip/hip_bf16.h>
#include <math.h>

// Problem constants (B,T,C)=(4,2048,2048), H=16, D=128
#define BB 4
#define TT 2048
#define CC 2048
#define HH 16
#define DD 128
#define MM (BB * TT)   // 8192 rows
#define N3 (3 * CC)    // 6144 qkv cols

typedef __bf16 bf16;
typedef __attribute__((ext_vector_type(16))) __bf16 v16bf;
typedef __attribute__((ext_vector_type(8)))  float  v8f;

#define WMMA_BF16(a, b, c) \
  __builtin_amdgcn_wmma_f32_16x16x32_bf16(false, (a), false, (b), (short)0, (c), false, false)

// ---- CDNA5 async copy: global -> LDS, 16B per lane, tracked by ASYNCcnt ----
static __device__ __forceinline__ void async_copy16(void* lds, const void* g) {
  asm volatile("global_load_async_to_lds_b128 %0, %1, off"
               :: "v"((unsigned)(uintptr_t)lds),
                  "v"((unsigned long long)(uintptr_t)g)
               : "memory");
}
static __device__ __forceinline__ void wait_async0() {
  asm volatile("s_wait_asynccnt 0x0" ::: "memory");
}

// ---- WMMA fragment loaders (layouts per CDNA5 ISA 7.12.2, wave32) ----
// A (16x32 bf16): lane L holds row M=L&15; K = {kb..kb+7, kb+16..kb+23}, kb=(L>>4)*8
static __device__ __forceinline__ v16bf load_frag_a(const bf16* base, int ld, int lane) {
  const int row = lane & 15;
  const int kb  = (lane >> 4) << 3;
  const bf16* p = base + row * ld + kb;
  union { v16bf v; uint4 u[2]; } r;
  r.u[0] = *reinterpret_cast<const uint4*>(p);
  r.u[1] = *reinterpret_cast<const uint4*>(p + 16);
  return r.v;
}
// B (32x16 bf16): lane L holds col N=L&15; K = {kb..kb+7, kb+16..kb+23}, kb=(L>>4)*8.
// `base` points at a row-major [N][K] tile (W rows / K rows / Vt rows).
static __device__ __forceinline__ v16bf load_frag_b(const bf16* base, int ld, int lane) {
  const int n  = lane & 15;
  const int kb = (lane >> 4) << 3;
  const bf16* p = base + n * ld + kb;
  union { v16bf v; uint4 u[2]; } r;
  r.u[0] = *reinterpret_cast<const uint4*>(p);
  r.u[1] = *reinterpret_cast<const uint4*>(p + 16);
  return r.v;
}

// ---------------- fp32 -> bf16 bulk convert (8 elems / thread) ----------------
__global__ void mha1250_cvt_bf16(const float* __restrict__ src, bf16* __restrict__ dst, int n8) {
  int i = blockIdx.x * blockDim.x + threadIdx.x;
  if (i >= n8) return;
  const float4* s = reinterpret_cast<const float4*>(src) + (size_t)i * 2;
  float4 a = s[0], b = s[1];
  union { bf16 h[8]; uint4 u; } o;
  o.h[0] = (bf16)a.x; o.h[1] = (bf16)a.y; o.h[2] = (bf16)a.z; o.h[3] = (bf16)a.w;
  o.h[4] = (bf16)b.x; o.h[5] = (bf16)b.y; o.h[6] = (bf16)b.z; o.h[7] = (bf16)b.w;
  reinterpret_cast<uint4*>(dst)[i] = o.u;
}

// ---- Wqkv prep: fold constant-angle RoPE (ref uses t=T for every position!)
// and the 1/sqrt(D) score scale into the q/k weight rows + bias. ----
__global__ void mha1250_prep_wqkv(const float* __restrict__ W, const float* __restrict__ bias,
                                  bf16* __restrict__ Wr, float* __restrict__ br) {
  const int row = blockIdx.x;          // 0..N3-1
  const int sec = row >> 11;           // 0=q 1=k 2=v
  const int cc  = row & (CC - 1);
  const int j   = cc & (DD - 1);       // dim within head
  float cj = 1.f, sj = 0.f, sgn = 1.f;
  int   prow = row;
  const float ascale = (sec == 0) ? 0.08838834764831845f : 1.0f; // 1/sqrt(128)
  if (sec < 2) {
    int i = j & 63;
    double ang = 2048.0 * pow(10000.0, -(double)(2 * i) / 128.0);
    cj = (float)cos(ang);
    sj = (float)sin(ang);
    int pj = (j < 64) ? (2 * j + 1) : (2 * (j - 64));
    sgn = (j < 64) ? -1.f : 1.f;
    prow = row - j + pj;
  }
  const float* w0 = W + (size_t)row * CC;
  const float* w1 = W + (size_t)prow * CC;
  bf16* out = Wr + (size_t)row * CC;
#pragma unroll
  for (int u = 0; u < CC / 256; ++u) {
    int col = threadIdx.x + u * 256;
    float v = cj * w0[col] + sgn * sj * w1[col];
    out[col] = (bf16)(v * ascale);
  }
  if (threadIdx.x == 0)
    br[row] = ascale * (cj * bias[row] + sgn * sj * bias[prow]);
}

// ---- shared GEMM core: async double-buffered staging, one barrier per k-step ----
// As/Bs tile: 128 rows x 32 k, row stride 40 (bank-conflict-free fragment reads).
static __device__ __forceinline__ void gemm_stage_async(
    const bf16* __restrict__ A, const bf16* __restrict__ Bw,
    bf16* As, bf16* Bs, int bm, int bn, int k0, int tid) {
#pragma unroll
  for (int u = 0; u < 2; ++u) {           // 512 16B chunks per tile
    int q = tid + u * 256;
    int r = q >> 2, c = (q & 3) << 3;
    async_copy16(&As[r * 40 + c], &A [(size_t)(bm + r) * CC + k0 + c]);
    async_copy16(&Bs[r * 40 + c], &Bw[(size_t)(bn + r) * CC + k0 + c]);
  }
}

static __device__ __forceinline__ void gemm_core(
    const bf16* __restrict__ A, const bf16* __restrict__ Bw,
    bf16 (*As)[128 * 40], bf16 (*Bs)[128 * 40],
    int bm, int bn, int tid, int lane, int wr, int wc, v8f acc[2][4]) {
#pragma unroll
  for (int mt = 0; mt < 2; ++mt)
#pragma unroll
    for (int nt = 0; nt < 4; ++nt)
#pragma unroll
      for (int e = 0; e < 8; ++e) acc[mt][nt][e] = 0.f;

  gemm_stage_async(A, Bw, As[0], Bs[0], bm, bn, 0, tid);

  for (int k0 = 0; k0 < CC; k0 += 32) {
    const int cur = (k0 >> 5) & 1, nxt = cur ^ 1;
    wait_async0();       // buf[cur] staged (and, transitively, visible after barrier)
    __syncthreads();     // also: everyone done reading buf[nxt] last iteration
    if (k0 + 32 < CC)
      gemm_stage_async(A, Bw, As[nxt], Bs[nxt], bm, bn, k0 + 32, tid);

    v16bf af[2], bfr[4];
#pragma unroll
    for (int mt = 0; mt < 2; ++mt)
      af[mt] = load_frag_a(&As[cur][(wr * 32 + mt * 16) * 40], 40, lane);
#pragma unroll
    for (int nt = 0; nt < 4; ++nt)
      bfr[nt] = load_frag_b(&Bs[cur][(wc * 64 + nt * 16) * 40], 40, lane);
#pragma unroll
    for (int mt = 0; mt < 2; ++mt)
#pragma unroll
      for (int nt = 0; nt < 4; ++nt)
        acc[mt][nt] = WMMA_BF16(af[mt], bfr[nt], acc[mt][nt]);
  }
}

// ---------------- QKV GEMM: [8192 x 6144] = xb @ Wrot^T + brot ----------------
// Epilogue: sec/h/b/t0 are uniform per block (128-wide tiles, 2048-wide sections,
// D=128, T=2048 all divide evenly) -> no per-element branching.
__global__ __launch_bounds__(256)
void mha1250_gemm_qkv(const bf16* __restrict__ X, const bf16* __restrict__ W,
                      const float* __restrict__ br,
                      bf16* __restrict__ Q, bf16* __restrict__ Kd, bf16* __restrict__ Vt) {
  __shared__ bf16 As[2][128 * 40];
  __shared__ bf16 Bs[2][128 * 40];
  const int tid = threadIdx.x, lane = tid & 31, w = tid >> 5;
  const int wr = w >> 1, wc = w & 1;
  const int bm = blockIdx.y * 128, bn = blockIdx.x * 128;

  v8f acc[2][4];
  gemm_core(X, W, As, Bs, bm, bn, tid, lane, wr, wc, acc);

  const int hl = lane >> 4, nl = lane & 15;
  const int sec = bn >> 11;                 // uniform: 0=q 1=k 2=v
  const int h   = (bn & (CC - 1)) >> 7;     // uniform head
  const int b   = bm >> 11;                 // uniform batch
  const int t0  = bm & (TT - 1);
  const size_t bh = (size_t)(b * HH + h);

  float bb[4];
#pragma unroll
  for (int nt = 0; nt < 4; ++nt) bb[nt] = br[bn + wc * 64 + nt * 16 + nl];

  if (sec < 2) {
    bf16* dst = sec ? Kd : Q;
#pragma unroll
    for (int mt = 0; mt < 2; ++mt)
#pragma unroll
      for (int nt = 0; nt < 4; ++nt)
#pragma unroll
        for (int e = 0; e < 8; ++e) {
          int t = t0 + wr * 32 + mt * 16 + (hl ? e + 8 : e);
          int d = wc * 64 + nt * 16 + nl;
          dst[(bh * TT + t) * DD + d] = (bf16)(acc[mt][nt][e] + bb[nt]);
        }
  } else {
#pragma unroll
    for (int mt = 0; mt < 2; ++mt)
#pragma unroll
      for (int nt = 0; nt < 4; ++nt)
#pragma unroll
        for (int e = 0; e < 8; ++e) {
          int t = t0 + wr * 32 + mt * 16 + (hl ? e + 8 : e);
          int d = wc * 64 + nt * 16 + nl;
          Vt[(bh * DD + d) * TT + t] = (bf16)(acc[mt][nt][e] + bb[nt]);
        }
  }
}

// ---------------- Flash attention: one (b,h) x 128 query rows per block ----------------
// Double-buffered K/V tiles: one wait_async+barrier per kv block, staging of the
// next block overlaps S=QK^T / softmax / PV of the current block.
__global__ __launch_bounds__(256)
void mha1250_attn(const bf16* __restrict__ Q, const bf16* __restrict__ K,
                  const bf16* __restrict__ Vt, bf16* __restrict__ Y) {
  __shared__ bf16 Qs[128 * 136];
  __shared__ bf16 Ks[2][64 * 136];
  __shared__ bf16 Vs[2][128 * 72];  // Vt tile: [d=128][kv=64]
  __shared__ bf16 Ps[128 * 72];     // probs, wave-private 16-row strips

  const int tid = threadIdx.x, lane = tid & 31, w = tid >> 5;
  const int qblk = blockIdx.x, bh = blockIdx.y;
  const int b = bh / HH, h = bh % HH;
  const size_t qk_base = (size_t)bh * TT * DD;
  const int q0 = qblk * 128;
  const int hl = lane >> 4, nl = lane & 15;

  auto stage_kv = [&](int kv0, int buf) {
#pragma unroll
    for (int u = 0; u < 4; ++u) {           // K 64x128
      int qd = tid + u * 256;
      int r = qd >> 4, c = (qd & 15) << 3;
      async_copy16(&Ks[buf][r * 136 + c], &K[qk_base + (size_t)(kv0 + r) * DD + c]);
    }
#pragma unroll
    for (int u = 0; u < 4; ++u) {           // Vt 128x64
      int qd = tid + u * 256;
      int r = qd >> 3, c = (qd & 7) << 3;
      async_copy16(&Vs[buf][r * 72 + c], &Vt[((size_t)bh * DD + r) * TT + kv0 + c]);
    }
  };

#pragma unroll
  for (int u = 0; u < 8; ++u) {             // stage Q 128x128 (async)
    int qd = tid + u * 256;
    int r = qd >> 4, c = (qd & 15) << 3;
    async_copy16(&Qs[r * 136 + c], &Q[qk_base + (size_t)(q0 + r) * DD + c]);
  }
  stage_kv(0, 0);

  float m_i[8], l_i[8];
  v8f oacc[8];
#pragma unroll
  for (int v = 0; v < 8; ++v) {
    m_i[v] = -1e30f; l_i[v] = 0.f;
#pragma unroll
    for (int e = 0; e < 8; ++e) oacc[v][e] = 0.f;
  }

  const int qr0 = q0 + w * 16;
  const int nkv = 2 * (qblk + 1);
  for (int j = 0; j < nkv; ++j) {
    const int kv0 = j * 64;
    const int cur = j & 1;
    wait_async0();       // buf[cur] (and Q on j==0) staged
    __syncthreads();     // visible to all; all done reading buf[cur^1]
    if (j + 1 < nkv) stage_kv(kv0 + 64, cur ^ 1);

    // S = Q Kt : 16x64 per wave
    v8f sacc[4];
#pragma unroll
    for (int nt = 0; nt < 4; ++nt)
#pragma unroll
      for (int e = 0; e < 8; ++e) sacc[nt][e] = 0.f;
#pragma unroll
    for (int k0 = 0; k0 < DD; k0 += 32) {
      v16bf aq = load_frag_a(&Qs[(w * 16) * 136 + k0], 136, lane);
#pragma unroll
      for (int nt = 0; nt < 4; ++nt) {
        v16bf bk = load_frag_b(&Ks[cur][(nt * 16) * 136 + k0], 136, lane);
        sacc[nt] = WMMA_BF16(aq, bk, sacc[nt]);
      }
    }

    if (kv0 + 63 > qr0) {                   // causal mask (diagonal region)
#pragma unroll
      for (int nt = 0; nt < 4; ++nt)
#pragma unroll
        for (int e = 0; e < 8; ++e) {
          int kvg = kv0 + nt * 16 + nl;
          int qg  = qr0 + (hl ? e + 8 : e);
          if (kvg > qg) sacc[nt][e] = -1e30f;
        }
    }

    // online softmax per row (row = VGPR slot e, cols in lanes)
#pragma unroll
    for (int e = 0; e < 8; ++e) {
      float x = fmaxf(fmaxf(sacc[0][e], sacc[1][e]), fmaxf(sacc[2][e], sacc[3][e]));
      x = fmaxf(x, __shfl_xor(x, 1, 32));
      x = fmaxf(x, __shfl_xor(x, 2, 32));
      x = fmaxf(x, __shfl_xor(x, 4, 32));
      x = fmaxf(x, __shfl_xor(x, 8, 32));
      float m_new = fmaxf(m_i[e], x);
      float alpha = __expf(m_i[e] - m_new);
      float rsum = 0.f;
      int prow = w * 16 + (hl ? e + 8 : e);
#pragma unroll
      for (int nt = 0; nt < 4; ++nt) {
        float p = __expf(sacc[nt][e] - m_new);
        rsum += p;
        Ps[prow * 72 + nt * 16 + nl] = (bf16)p;
      }
      rsum += __shfl_xor(rsum, 1, 32);
      rsum += __shfl_xor(rsum, 2, 32);
      rsum += __shfl_xor(rsum, 4, 32);
      rsum += __shfl_xor(rsum, 8, 32);
      l_i[e] = l_i[e] * alpha + rsum;
      m_i[e] = m_new;
#pragma unroll
      for (int nt2 = 0; nt2 < 8; ++nt2) oacc[nt2][e] *= alpha;
    }

    // O += P @ V  (A = P 16x64, B cols contiguous in Vt rows)
#pragma unroll
    for (int k0 = 0; k0 < 64; k0 += 32) {
      v16bf ap = load_frag_a(&Ps[(w * 16) * 72 + k0], 72, lane);
#pragma unroll
      for (int nt2 = 0; nt2 < 8; ++nt2) {
        v16bf bv = load_frag_b(&Vs[cur][(nt2 * 16) * 72 + k0], 72, lane);
        oacc[nt2] = WMMA_BF16(ap, bv, oacc[nt2]);
      }
    }
  }

#pragma unroll
  for (int e = 0; e < 8; ++e) {
    float inv = 1.f / l_i[e];
    int t = q0 + w * 16 + (hl ? e + 8 : e);
    size_t rowbase = ((size_t)b * TT + t) * CC + (size_t)h * DD;
#pragma unroll
    for (int nt2 = 0; nt2 < 8; ++nt2)
      Y[rowbase + nt2 * 16 + nl] = (bf16)(oacc[nt2][e] * inv);
  }
}

// ---------------- Output projection: out = Y @ Wout^T + bout (fp32 out) ----------------
__global__ __launch_bounds__(256)
void mha1250_gemm_out(const bf16* __restrict__ Yb, const bf16* __restrict__ W,
                      const float* __restrict__ bias, float* __restrict__ out) {
  __shared__ bf16 As[2][128 * 40];
  __shared__ bf16 Bs[2][128 * 40];
  const int tid = threadIdx.x, lane = tid & 31, w = tid >> 5;
  const int wr = w >> 1, wc = w & 1;
  const int bm = blockIdx.y * 128, bn = blockIdx.x * 128;

  v8f acc[2][4];
  gemm_core(Yb, W, As, Bs, bm, bn, tid, lane, wr, wc, acc);

  const int hl = lane >> 4, nl = lane & 15;
  float bb[4];
#pragma unroll
  for (int nt = 0; nt < 4; ++nt) bb[nt] = bias[bn + wc * 64 + nt * 16 + nl];
#pragma unroll
  for (int mt = 0; mt < 2; ++mt)
#pragma unroll
    for (int nt = 0; nt < 4; ++nt)
#pragma unroll
      for (int e = 0; e < 8; ++e) {
        int gm = bm + wr * 32 + mt * 16 + (hl ? e + 8 : e);
        int gn = bn + wc * 64 + nt * 16 + nl;
        out[(size_t)gm * CC + gn] = acc[mt][nt][e] + bb[nt];
      }
}

// ---------------- host launch ----------------
extern "C" void kernel_launch(void* const* d_in, const int* in_sizes, int n_in,
                              void* d_out, int out_size, void* d_ws, size_t ws_size,
                              hipStream_t stream) {
  const float* x    = (const float*)d_in[0];
  const float* Wqkv = (const float*)d_in[1];
  const float* bqkv = (const float*)d_in[2];
  const float* Wout = (const float*)d_in[3];
  const float* bout = (const float*)d_in[4];

  char* ws = (char*)d_ws;
  bf16*  xb = (bf16*)(ws);                         // 33,554,432 B
  bf16*  wq = (bf16*)(ws + 33554432ull);           // 25,165,824 B
  bf16*  wo = (bf16*)(ws + 58720256ull);           //  8,388,608 B
  float* br = (float*)(ws + 67108864ull);          //     24,576 B
  bf16*  Qb = (bf16*)(ws + 67133440ull);           // 33,554,432 B
  bf16*  Kb = (bf16*)(ws + 100687872ull);          // 33,554,432 B
  bf16*  Vt = (bf16*)(ws + 134242304ull);          // 33,554,432 B
  bf16*  Yb = (bf16*)(ws + 167796736ull);          // 33,554,432 B (ends ~201.4 MB)

  mha1250_cvt_bf16<<<(MM * CC / 8 + 255) / 256, 256, 0, stream>>>(x, xb, MM * CC / 8);
  mha1250_cvt_bf16<<<(CC * CC / 8 + 255) / 256, 256, 0, stream>>>(Wout, wo, CC * CC / 8);
  mha1250_prep_wqkv<<<N3, 256, 0, stream>>>(Wqkv, bqkv, wq, br);
  mha1250_gemm_qkv<<<dim3(N3 / 128, MM / 128), 256, 0, stream>>>(xb, wq, br, Qb, Kb, Vt);
  mha1250_attn<<<dim3(TT / 128, BB * HH), 256, 0, stream>>>(Qb, Kb, Vt, Yb);
  mha1250_gemm_out<<<dim3(CC / 128, MM / 128), 256, 0, stream>>>(Yb, wo, bout, (float*)d_out);
}
